// RelNet_6691559047523
// MI455X (gfx1250) — compile-verified, matrix operand order checked by reading, not run on hardware
//
#include <hip/hip_runtime.h>
#include <hip/hip_bf16.h>
#include <cstdint>
#include <cstddef>

// ---------------- problem constants ----------------
#define E_EDGES   131072
#define N_NODES   8192
#define G_GRAPHS  64
#define ED        2049
#define KF        2055        // ED + 6 (concat of endpoint coords)
#define KF_PAD    2080        // K rounded to multiple of 32 for bf16 WMMA
#define EMB       625
#define EMB_PAD   640
#define HID       128
#define OUTD      2
#define BN_EPS    1e-5f

// ---------------- tiling ----------------
#define BM      128           // edge rows per workgroup
#define KB      32            // K depth per WMMA step (16x16x32 bf16)
#define NTHREADS 512          // 16 wave32: 2 row-groups x 8 col-groups
#define NT      5             // 16-wide col tiles per wave (8*5*16 = 640)
#define MT      4             // 16-tall row tiles per wave (4*16 = 64)
#define KSTEPS1 (KF_PAD / KB) // 65
#define KSTEPS2 (EMB_PAD / KB)// 20

// ---------------- LDS layout (exactly 320 KB, dynamic base = 0) ----------------
// region1 [0, 163840):        h1 (128x640 bf16). Dead during layer 3 ->
//                             hosts layer-3 B-stage double buffer.
// region2 [163840, 327680):   h2 (128x640 bf16). Dead during layers 1-2 ->
//                             hosts B-stage (2x40KB) + A-stage (2x8KB) + coords.
#define REG2_OFF      163840
#define BSTAGE_ELEMS  (KB * EMB_PAD)       // 20480 bf16
#define BSTAGE_BYTES  (BSTAGE_ELEMS * 2)   // 40 KB
#define ASTAGE_ELEMS  (BM * KB)            // 4096 bf16 = 8 KB
#define LDS_TOTAL     327680

typedef __attribute__((ext_vector_type(16))) __bf16 v16bf;
typedef __attribute__((ext_vector_type(8)))  __bf16 v8bf;
typedef __attribute__((ext_vector_type(8)))  float  v8f;

// Per-lane 16x32-bf16 fragment loader.
// 16-bit A layout: lanes 0-15 hold row M=lane with K {0..7, 16..23},
// lanes 16-31 hold row M=lane-16 with K {8..15, 24..31}. Caller passes
// p = row_base + (lane>>4)*8; we read [p, p+8) and [p+16, p+24).
__device__ __forceinline__ v16bf load_frag(const __bf16* p) {
  v8bf lo = *(const v8bf*)(p);
  v8bf hi = *(const v8bf*)(p + 16);
  v16bf r;
#pragma unroll
  for (int i = 0; i < 8; ++i) { r[i] = lo[i]; r[8 + i] = hi[i]; }
  return r;
}

__device__ __forceinline__ void zero_acc(v8f acc[MT][NT]) {
#pragma unroll
  for (int m = 0; m < MT; ++m)
#pragma unroll
    for (int n = 0; n < NT; ++n)
#pragma unroll
      for (int j = 0; j < 8; ++j) acc[m][n][j] = 0.f;
}

// ---- async B k-slice staging: 32x640 bf16 copied straight into LDS via the
// CDNA5 async-to-LDS engine (ASYNCcnt), no VGPR round-trip. Stored [n][32] so
// fragment reads are two contiguous ds_load_b128 per lane. ----
__device__ __forceinline__ void asyncB_slice(const __bf16* __restrict__ Wt, int ldk,
                                             int k0, uint32_t lds_base_bytes, int tid) {
#pragma unroll
  for (int r = 0; r < 5; ++r) {
    int idx = r * NTHREADS + tid;         // 2560 chunks of 16B = 40 KB
    int n = idx >> 2, kc = idx & 3;
    uint64_t ga = (uint64_t)(uintptr_t)(Wt + (size_t)n * ldk + k0 + kc * 8);
    uint32_t la = lds_base_bytes + (uint32_t)((n * KB + kc * 8) * 2);
    asm volatile("global_load_async_to_lds_b128 %0, %1, off"
                 :: "v"(la), "v"(ga) : "memory");
  }
}
__device__ __forceinline__ void wait_async_all() {
  asm volatile("s_wait_asynccnt 0x0" ::: "memory");
}

// ---- layer-1 A k-slice staging: 128x32 f32 -> bf16 (needs conversion, so it
// goes through VGPRs). lane = column so every global load is a fully
// coalesced 128B line; the branch is thread-uniform except in the tail slice.
__device__ __forceinline__ void stageA_load(float t[8], const float* __restrict__ edge_attr,
                                            const float* c6, int e0, int k0, int tid) {
  int col  = tid & 31;
  int r0   = (tid >> 5) * 8;
  int cidx = k0 + col;
#pragma unroll
  for (int i = 0; i < 8; ++i) {
    int row = r0 + i;
    float v;
    if (cidx < ED)      v = edge_attr[(size_t)(e0 + row) * ED + cidx];
    else if (cidx < KF) v = c6[row * 6 + (cidx - ED)];
    else                v = 0.f;
    t[i] = v;
  }
}
__device__ __forceinline__ void stageA_store(const float t[8], __bf16* dst, int tid) {
  int col = tid & 31;
  int r0  = (tid >> 5) * 8;
#pragma unroll
  for (int i = 0; i < 8; ++i) dst[(r0 + i) * KB + col] = (__bf16)t[i];
}

// ---- one 32-deep WMMA step: A frags from LDS activations, B frags from the
// staged LDS k-slice. ----
__device__ __forceinline__ void mma_step(v8f acc[MT][NT], const __bf16* asrc, int lda,
                                         int koff, const __bf16* bbuf,
                                         int m_base, int n_base, int nlane, int half) {
  v16bf af[MT];
#pragma unroll
  for (int m = 0; m < MT; ++m)
    af[m] = load_frag(asrc + (m_base + m * 16 + nlane) * lda + koff + half * 8);
#pragma unroll
  for (int n = 0; n < NT; ++n) {
    v16bf bfrag = load_frag(bbuf + (n_base + n * 16 + nlane) * KB + half * 8);
#pragma unroll
    for (int m = 0; m < MT; ++m)
      acc[m][n] = __builtin_amdgcn_wmma_f32_16x16x32_bf16(
          false, af[m], false, bfrag, (short)0, acc[m][n], false, false);
  }
}

// ---- full K loop for layers 2/3: A resident in LDS, B double-buffer staged
// by the async engine; software pipelined (issue async slice s+1 -> WMMA on
// slice s -> s_wait_asynccnt -> barrier). ----
__device__ __forceinline__ void gemm_layer(v8f acc[MT][NT], const __bf16* hsrc,
                                           const __bf16* __restrict__ Wt,
                                           const __bf16* bstage_ptr, uint32_t bstage_off,
                                           int m_base, int n_base, int nlane, int half,
                                           int tid) {
  asyncB_slice(Wt, EMB_PAD, 0, bstage_off, tid);
  wait_async_all();
  __syncthreads();
  for (int s = 0; s < KSTEPS2; ++s) {
    int p = s & 1;
    if (s + 1 < KSTEPS2)
      asyncB_slice(Wt, EMB_PAD, (s + 1) * KB, bstage_off + (p ^ 1) * BSTAGE_BYTES, tid);
    mma_step(acc, hsrc, EMB_PAD, s * KB, bstage_ptr + p * BSTAGE_ELEMS,
             m_base, n_base, nlane, half);
    wait_async_all();
    __syncthreads();
  }
}

// Epilogue: y = a[col]*ReLU(acc + b[col]) + c[col] -> bf16 LDS tile.
// C layout: VGPR j -> row m*16 + j + 8*(lane>>4); col = n*16 + (lane&15).
__device__ __forceinline__ void epilogue_lds(v8f acc[MT][NT], __bf16* hdst,
                                             const float* __restrict__ b,
                                             const float* __restrict__ a,
                                             const float* __restrict__ c,
                                             int m_base, int n_base, int nlane, int half) {
#pragma unroll
  for (int n = 0; n < NT; ++n) {
    int col = n_base + n * 16 + nlane;
    float bb = b[col], aa = a[col], cc = c[col];
#pragma unroll
    for (int m = 0; m < MT; ++m)
#pragma unroll
      for (int j = 0; j < 8; ++j) {
        int row = m_base + m * 16 + half * 8 + j;
        float v = aa * fmaxf(acc[m][n][j] + bb, 0.f) + cc;
        hdst[row * EMB_PAD + col] = (__bf16)v;
      }
  }
}

// Final-layer epilogue: affine then atomic segment-sum into sums[g][col].
__device__ __forceinline__ void epilogue_atomic(v8f acc[MT][NT], float* __restrict__ sums,
                                                const int* __restrict__ ebatch_e0,
                                                const float* __restrict__ b,
                                                const float* __restrict__ a,
                                                const float* __restrict__ c,
                                                int m_base, int n_base, int nlane, int half) {
  float bb[NT], aa[NT], cc[NT];
  bool ok[NT];
#pragma unroll
  for (int n = 0; n < NT; ++n) {
    int col = n_base + n * 16 + nlane;
    ok[n] = col < EMB;
    bb[n] = b[col]; aa[n] = a[col]; cc[n] = c[col];
  }
#pragma unroll
  for (int m = 0; m < MT; ++m)
#pragma unroll
    for (int j = 0; j < 8; ++j) {
      int row = m_base + m * 16 + half * 8 + j;
      int g = ebatch_e0[row];           // L1-hot: 128 values reused block-wide
#pragma unroll
      for (int n = 0; n < NT; ++n)
        if (ok[n]) {
          int col = n_base + n * 16 + nlane;
          float v = aa[n] * fmaxf(acc[m][n][j] + bb[n], 0.f) + cc[n];
          atomicAdd(&sums[(size_t)g * EMB_PAD + col], v);
        }
    }
}

// ------------------------------------------------------------------
// Fused 3-layer edge MLP. 1024 blocks x 512 threads (16 wave32).
// ------------------------------------------------------------------
__global__ __launch_bounds__(NTHREADS, 1)
void relnet_main(const float* __restrict__ edge_attr,
                 const float* __restrict__ array,
                 const int* __restrict__ ei_src,
                 const int* __restrict__ ei_dst,
                 const int* __restrict__ ebatch,
                 const __bf16* __restrict__ W1t,   // [EMB_PAD][KF_PAD]
                 const __bf16* __restrict__ W2t,   // [EMB_PAD][EMB_PAD]
                 const __bf16* __restrict__ W3t,   // [EMB_PAD][EMB_PAD]
                 const float* __restrict__ aff,    // 9 x EMB_PAD
                 float* __restrict__ sums)         // [G][EMB_PAD]
{
  extern __shared__ char smem[];
  __bf16* h1    = (__bf16*)smem;                                        // 160 KB
  __bf16* h2    = (__bf16*)(smem + REG2_OFF);                           // 160 KB
  __bf16* bst12 = (__bf16*)(smem + REG2_OFF);                           // 2 x 40 KB
  __bf16* hA    = (__bf16*)(smem + REG2_OFF + 2 * BSTAGE_BYTES);        // 2 x 8 KB
  float*  c6    = (float*)(smem + REG2_OFF + 2 * BSTAGE_BYTES
                                           + 2 * ASTAGE_ELEMS * 2);     // 3 KB
  __bf16* bst3  = (__bf16*)smem;                                        // 2 x 40 KB

  const int tid    = threadIdx.x;
  const int lane   = tid & 31;
  const int wave   = tid >> 5;                 // 0..15
  const int nlane  = lane & 15;
  const int half   = lane >> 4;
  const int m_base = (wave >> 3) * 64;         // row-group: 0 or 64
  const int n_base = (wave & 7) * (NT * 16);   // col-group: 0..560
  const int e0     = blockIdx.x * BM;

  // Stage endpoint coords for the layer-1 tail slice.
  if (tid < BM) {
    int e = e0 + tid;
    int s = ei_src[e];
    int d = ei_dst[e];
#pragma unroll
    for (int i = 0; i < 3; ++i) {
      c6[tid * 6 + i]     = array[s * 3 + i];
      c6[tid * 6 + 3 + i] = array[d * 3 + i];
    }
  }
  __syncthreads();

  v8f acc[MT][NT];
  zero_acc(acc);

  // ---------------- layer 1: [128 x 2080] x [2080 x 640] ----------------
  {
    float ta[8];
    asyncB_slice(W1t, KF_PAD, 0, REG2_OFF, tid);
    stageA_load(ta, edge_attr, c6, e0, 0, tid);
    stageA_store(ta, hA, tid);
    wait_async_all();
    __syncthreads();
    for (int s = 0; s < KSTEPS1; ++s) {
      int p = s & 1;
      if (s + 1 < KSTEPS1) {
        asyncB_slice(W1t, KF_PAD, (s + 1) * KB, REG2_OFF + (p ^ 1) * BSTAGE_BYTES, tid);
        stageA_load(ta, edge_attr, c6, e0, (s + 1) * KB, tid);
      }
      mma_step(acc, hA + p * ASTAGE_ELEMS, KB, 0, bst12 + p * BSTAGE_ELEMS,
               m_base, n_base, nlane, half);
      if (s + 1 < KSTEPS1) stageA_store(ta, hA + (p ^ 1) * ASTAGE_ELEMS, tid);
      wait_async_all();
      __syncthreads();
    }
  }
  epilogue_lds(acc, h1, aff + 0 * EMB_PAD, aff + 1 * EMB_PAD, aff + 2 * EMB_PAD,
               m_base, n_base, nlane, half);
  __syncthreads();

  // ---------------- layer 2 (A = h1, async-stage B in dead h2 region) --------
  zero_acc(acc);
  gemm_layer(acc, h1, W2t, bst12, REG2_OFF, m_base, n_base, nlane, half, tid);
  epilogue_lds(acc, h2, aff + 3 * EMB_PAD, aff + 4 * EMB_PAD, aff + 5 * EMB_PAD,
               m_base, n_base, nlane, half);
  __syncthreads();

  // ---------------- layer 3 (A = h2, async-stage B in dead h1 region) + seg-sum
  zero_acc(acc);
  gemm_layer(acc, h2, W3t, bst3, 0u, m_base, n_base, nlane, half, tid);
  epilogue_atomic(acc, sums, ebatch + e0, aff + 6 * EMB_PAD, aff + 7 * EMB_PAD,
                  aff + 8 * EMB_PAD, m_base, n_base, nlane, half);
}

// ------------------------------------------------------------------
// Prep / reduction / head kernels
// ------------------------------------------------------------------
__global__ void k_pack_w1(const float* __restrict__ W1, __bf16* __restrict__ W1t) {
  int idx = blockIdx.x * 256 + threadIdx.x;
  if (idx >= EMB_PAD * KF_PAD) return;
  int n = idx / KF_PAD, k = idx - n * KF_PAD;
  float v = (n < EMB && k < KF) ? W1[(size_t)k * EMB + n] : 0.f;
  W1t[idx] = (__bf16)v;
}

__global__ void k_pack_w(const float* __restrict__ W, __bf16* __restrict__ Wt) {
  int idx = blockIdx.x * 256 + threadIdx.x;
  if (idx >= EMB_PAD * EMB_PAD) return;
  int n = idx / EMB_PAD, k = idx - n * EMB_PAD;
  float v = (n < EMB && k < EMB) ? W[(size_t)k * EMB + n] : 0.f;
  Wt[idx] = (__bf16)v;
}

// Fold BN (eval) into per-column affine: y = a*ReLU(x + b) + c
__global__ void k_affine(const float* __restrict__ b, const float* __restrict__ g,
                         const float* __restrict__ be, const float* __restrict__ m,
                         const float* __restrict__ v, float* __restrict__ out) {
  int i = blockIdx.x * 128 + threadIdx.x;
  if (i >= EMB_PAD) return;
  if (i < EMB) {
    float a = g[i] * rsqrtf(v[i] + BN_EPS);
    out[i]               = b[i];
    out[EMB_PAD + i]     = a;
    out[2 * EMB_PAD + i] = be[i] - m[i] * a;
  } else {  // padded cols feed exact zeros into the next layer's K dim
    out[i] = 0.f; out[EMB_PAD + i] = 0.f; out[2 * EMB_PAD + i] = 0.f;
  }
}

__global__ void k_init(float* __restrict__ sums, float* __restrict__ cnt) {
  int i = blockIdx.x * 256 + threadIdx.x;
  if (i < G_GRAPHS * EMB_PAD) sums[i] = 0.f;
  if (i < G_GRAPHS) cnt[i] = 0.f;
}

// ebatch[e] = batch[src[e]] (consumed by the fused kernel's epilogue) + counts.
__global__ void k_ebatch(const int* __restrict__ src, const int* __restrict__ batch,
                         int* __restrict__ ebatch, float* __restrict__ cnt) {
  int e = blockIdx.x * 256 + threadIdx.x;
  if (e < E_EDGES) {
    int g = batch[src[e]];
    ebatch[e] = g;
    atomicAdd(&cnt[g], 1.f);
  }
}

// Mean + 625->128->2 head. 64 blocks, tiny (f32 VALU is fine here).
__global__ __launch_bounds__(128)
void k_head(const float* __restrict__ sums, const float* __restrict__ cnt,
            const float* __restrict__ Wf1, const float* __restrict__ bf1,
            const float* __restrict__ Wf2, const float* __restrict__ bf2,
            float* __restrict__ out) {
  __shared__ float gf[EMB];
  __shared__ float hid[HID];
  int g = blockIdx.x, t = threadIdx.x;
  float inv = 1.f / fmaxf(cnt[g], 1.f);
  for (int c = t; c < EMB; c += 128) gf[c] = sums[(size_t)g * EMB_PAD + c] * inv;
  __syncthreads();
  float acc = bf1[t];
  for (int c = 0; c < EMB; ++c) acc += gf[c] * Wf1[(size_t)c * HID + t];
  hid[t] = fmaxf(acc, 0.f);
  __syncthreads();
  if (t < OUTD) {
    float o = bf2[t];
#pragma unroll 8
    for (int j = 0; j < HID; ++j) o += hid[j] * Wf2[(size_t)j * OUTD + t];
    out[g * OUTD + t] = o;
  }
}

// ------------------------------------------------------------------
extern "C" void kernel_launch(void* const* d_in, const int* in_sizes, int n_in,
                              void* d_out, int out_size, void* d_ws, size_t ws_size,
                              hipStream_t stream) {
  (void)in_sizes; (void)n_in; (void)out_size; (void)ws_size;

  const float* edge_attr = (const float*)d_in[0];
  const float* array_    = (const float*)d_in[1];
  const int*   ei        = (const int*)d_in[2];      // [2][E]
  const int*   batch     = (const int*)d_in[3];
  const float* W1 = (const float*)d_in[4];
  const float* b1 = (const float*)d_in[5];
  const float* g1 = (const float*)d_in[6];
  const float* be1= (const float*)d_in[7];
  const float* m1 = (const float*)d_in[8];
  const float* v1 = (const float*)d_in[9];
  const float* W2 = (const float*)d_in[10];
  const float* b2 = (const float*)d_in[11];
  const float* g2 = (const float*)d_in[12];
  const float* be2= (const float*)d_in[13];
  const float* m2 = (const float*)d_in[14];
  const float* v2 = (const float*)d_in[15];
  const float* W3 = (const float*)d_in[16];
  const float* b3 = (const float*)d_in[17];
  const float* g3 = (const float*)d_in[18];
  const float* be3= (const float*)d_in[19];
  const float* m3 = (const float*)d_in[20];
  const float* v3 = (const float*)d_in[21];
  const float* Wf1= (const float*)d_in[22];
  const float* bf1= (const float*)d_in[23];
  const float* Wf2= (const float*)d_in[24];
  const float* bf2= (const float*)d_in[25];

  const int* src = ei;
  const int* dst = ei + E_EDGES;

  // ---- workspace carve (256B-aligned) ----
  char* ws = (char*)d_ws;
  size_t off = 0;
  auto carve = [&](size_t bytes) -> void* {
    void* p = ws + off;
    off += (bytes + 255) & ~(size_t)255;
    return p;
  };
  __bf16* W1t   = (__bf16*)carve((size_t)EMB_PAD * KF_PAD * sizeof(__bf16));
  __bf16* W2t   = (__bf16*)carve((size_t)EMB_PAD * EMB_PAD * sizeof(__bf16));
  __bf16* W3t   = (__bf16*)carve((size_t)EMB_PAD * EMB_PAD * sizeof(__bf16));
  float*  aff   = (float*)carve((size_t)9 * EMB_PAD * sizeof(float));
  float*  sums  = (float*)carve((size_t)G_GRAPHS * EMB_PAD * sizeof(float));
  float*  cnt   = (float*)carve((size_t)G_GRAPHS * sizeof(float));
  int*    ebatch= (int*)carve((size_t)E_EDGES * sizeof(int));

  // ---- prep ----
  k_pack_w1<<<(EMB_PAD * KF_PAD + 255) / 256, 256, 0, stream>>>(W1, W1t);
  k_pack_w <<<(EMB_PAD * EMB_PAD + 255) / 256, 256, 0, stream>>>(W2, W2t);
  k_pack_w <<<(EMB_PAD * EMB_PAD + 255) / 256, 256, 0, stream>>>(W3, W3t);
  k_affine<<<(EMB_PAD + 127) / 128, 128, 0, stream>>>(b1, g1, be1, m1, v1, aff + 0 * EMB_PAD);
  k_affine<<<(EMB_PAD + 127) / 128, 128, 0, stream>>>(b2, g2, be2, m2, v2, aff + 3 * EMB_PAD);
  k_affine<<<(EMB_PAD + 127) / 128, 128, 0, stream>>>(b3, g3, be3, m3, v3, aff + 6 * EMB_PAD);
  k_init<<<(G_GRAPHS * EMB_PAD + 255) / 256, 256, 0, stream>>>(sums, cnt);
  k_ebatch<<<(E_EDGES + 255) / 256, 256, 0, stream>>>(src, batch, ebatch, cnt);

  // ---- fused 3-layer WMMA MLP + segment-sum ----
  relnet_main<<<E_EDGES / BM, NTHREADS, LDS_TOTAL, stream>>>(
      edge_attr, array_, src, dst, ebatch, W1t, W2t, W3t, aff, sums);

  // ---- mean + head ----
  k_head<<<G_GRAPHS, 128, 0, stream>>>(sums, cnt, Wf1, bf1, Wf2, bf2, (float*)d_out);
}
